// Multihead_Attention_7816840479254
// MI455X (gfx1250) — matrix operand order repaired and checked by reading, not compiled, not run on hardware
//
#include <hip/hip_runtime.h>

// ---------------------------------------------------------------------------
// Multihead "attention" (no softmax) for MI455X / gfx1250.
//
// Shapes: x,y:[2,2048,1024]  weights:[1024,1024]  biases:[1024]
// Key identities exploited:
//   * reshape(16,2,2048,64) on [2,2048,1024] => head-block g = h*2+b is the
//     CONTIGUOUS [2048 x 64] row-major slab at flat offset g*131072.
//   * no softmax => (Q K^T) V == Q (K^T V), with K^T V only 64x64 per block.
//
// Roofline: 4 big GEMMs (4096x1024x1024) = 34 GFLOP vs ~200MB HBM traffic
// (~9us @ 23.3TB/s). Pure-f32 WMMA (16x16x4, 1024 MAC/instr) is ~8x slower
// per instr than f16 WMMA (16x16x32) => f32 path is compute-bound (~115us).
// Fix: split-f16 GEMM: a = hi+lo (hi=f16(a), lo=f16(a-hi)); a*w ~= hi*hi +
// hi*lo + lo*hi. 3 f16 WMMAs replace 8 f32 WMMAs per 32-K chunk (2.7x) with
// ~2^-22 relative error (f32-faithful). Tiny attention stages (~1 GFLOP)
// stay on the exact f32 16x16x4 path.
// ---------------------------------------------------------------------------

typedef __attribute__((ext_vector_type(2)))  float     v2f;
typedef __attribute__((ext_vector_type(8)))  float     v8f;
typedef __attribute__((ext_vector_type(4)))  _Float16  v4h;
typedef __attribute__((ext_vector_type(8)))  _Float16  v8h;
typedef __attribute__((ext_vector_type(16))) _Float16  v16h;

__device__ __forceinline__ v8f wmma4(v2f a, v2f b, v8f c) {
  // D(16x16,f32) = A(16x4,f32) * B(4x16,f32) + C  -- v_wmma_f32_16x16x4_f32
  return __builtin_amdgcn_wmma_f32_16x16x4_f32(false, a, false, b, (short)0, c,
                                               false, false);
}
__device__ __forceinline__ v8f wmma32(v16h a, v16h b, v8f c) {
  // D(16x16,f32) = A(16x32,f16) * B(32x16,f16) + C -- v_wmma_f32_16x16x32_f16
  return __builtin_amdgcn_wmma_f32_16x16x32_f16(false, a, false, b, (short)0,
                                                c, false, false);
}
__device__ __forceinline__ v16h cat8(v8h p, v8h q) {
  return __builtin_shufflevector(p, q, 0, 1, 2, 3, 4, 5, 6, 7, 8, 9, 10, 11,
                                 12, 13, 14, 15);
}
// Split one float4 into hi/lo f16 quads and store to LDS.
__device__ __forceinline__ void split_store(float4 v, _Float16* hp,
                                            _Float16* lp) {
  const _Float16 h0 = (_Float16)v.x, h1 = (_Float16)v.y, h2 = (_Float16)v.z,
                 h3 = (_Float16)v.w;
  *(v4h*)hp = (v4h){h0, h1, h2, h3};
  *(v4h*)lp = (v4h){(_Float16)(v.x - (float)h0), (_Float16)(v.y - (float)h1),
                    (_Float16)(v.z - (float)h2), (_Float16)(v.w - (float)h3)};
}

// ---------------------------------------------------------------------------
// C[M,N] = A[M,K] @ W[N,K]^T + bias[N], split-f16 path.
// Grid: (M/128, N/64), block 256 (8 waves; wave tile 32x32, K-chunk 32).
// ISA 7.12.2 fragment layouts (16-bit, wave32):
//   A(16x32) lane L: row m0+(L&15); K = {8*lh..8*lh+7} U {16+8*lh..+7}
//   B(32x16) lane L: col n0+(L&15); K = {8*lh..8*lh+7} U {16+8*lh..+7}
//     B[k,n] = W[n,k] => W's row-major rows feed B columns contiguously.
//   C/D(16x16) vgpr r, lane L: C[m0 + 8*lh + r, n0 + (L&15)]
// ---------------------------------------------------------------------------
__global__ __launch_bounds__(256) void gemm_bias_bt_f16x2_kernel(
    const float* __restrict__ A, const float* __restrict__ W,
    const float* __restrict__ bias, float* __restrict__ C,
    const int K, const int N) {
  // pitch 40 halfs = 80B: rows 16B-aligned, b128 frag reads conflict-free.
  __shared__ __align__(16) _Float16 Ah[128][40];
  __shared__ __align__(16) _Float16 Al[128][40];
  __shared__ __align__(16) _Float16 Wh[64][40];
  __shared__ __align__(16) _Float16 Wl[64][40];

  const int tid  = threadIdx.x;
  const int lane = tid & 31;
  const int wave = tid >> 5;
  const int lh   = lane >> 4;
  const int l16  = lane & 15;
  const int wm   = (wave & 3) * 32;   // wave M offset in 128-row tile
  const int wn   = (wave >> 2) * 32;  // wave N offset in 64-col tile
  const int m0   = blockIdx.x * 128;
  const int n0   = blockIdx.y * 64;

  // Staging coordinates: each thread owns (row r in a 32-row group, 4 cols).
  const int sr = tid >> 3;          // 0..31
  const int sc = (tid & 7) * 4;     // 0,4,...,28

  v8f acc[2][2] = {};

  for (int kb = 0; kb < K; kb += 32) {
    // ---- issue ALL six staging loads first (max per-wave MLP), ----------
    // ---- then convert+store; loadcnt waits retire progressively. --------
    const float4 va0 = *(const float4*)(A + (size_t)(m0 + sr) * K + kb + sc);
    const float4 va1 =
        *(const float4*)(A + (size_t)(m0 + 32 + sr) * K + kb + sc);
    const float4 va2 =
        *(const float4*)(A + (size_t)(m0 + 64 + sr) * K + kb + sc);
    const float4 va3 =
        *(const float4*)(A + (size_t)(m0 + 96 + sr) * K + kb + sc);
    const float4 vw0 = *(const float4*)(W + (size_t)(n0 + sr) * K + kb + sc);
    const float4 vw1 =
        *(const float4*)(W + (size_t)(n0 + 32 + sr) * K + kb + sc);

    if (kb + 32 < K) {  // prefetch next K-chunk (global_prefetch_b8)
      __builtin_prefetch(A + (size_t)(m0 + sr) * K + kb + 32 + sc, 0, 0);
      __builtin_prefetch(W + (size_t)(n0 + sr) * K + kb + 32 + sc, 0, 0);
    }

    split_store(va0, &Ah[sr][sc], &Al[sr][sc]);
    split_store(va1, &Ah[32 + sr][sc], &Al[32 + sr][sc]);
    split_store(va2, &Ah[64 + sr][sc], &Al[64 + sr][sc]);
    split_store(va3, &Ah[96 + sr][sc], &Al[96 + sr][sc]);
    split_store(vw0, &Wh[sr][sc], &Wl[sr][sc]);
    split_store(vw1, &Wh[32 + sr][sc], &Wl[32 + sr][sc]);
    __syncthreads();

    // ---- fragments ------------------------------------------------------
    const int ka = 8 * lh;
    v16h ah0 = cat8(*(const v8h*)&Ah[wm + l16][8 * lh],
                    *(const v8h*)&Ah[wm + l16][16 + 8 * lh]);
    v16h ah1 = cat8(*(const v8h*)&Ah[wm + 16 + l16][8 * lh],
                    *(const v8h*)&Ah[wm + 16 + l16][16 + 8 * lh]);
    v16h al0 = cat8(*(const v8h*)&Al[wm + l16][8 * lh],
                    *(const v8h*)&Al[wm + l16][16 + 8 * lh]);
    v16h al1 = cat8(*(const v8h*)&Al[wm + 16 + l16][8 * lh],
                    *(const v8h*)&Al[wm + 16 + l16][16 + 8 * lh]);
    v16h bh0 = cat8(*(const v8h*)&Wh[wn + l16][ka],
                    *(const v8h*)&Wh[wn + l16][ka + 16]);
    v16h bh1 = cat8(*(const v8h*)&Wh[wn + 16 + l16][ka],
                    *(const v8h*)&Wh[wn + 16 + l16][ka + 16]);
    v16h bl0 = cat8(*(const v8h*)&Wl[wn + l16][ka],
                    *(const v8h*)&Wl[wn + l16][ka + 16]);
    v16h bl1 = cat8(*(const v8h*)&Wl[wn + 16 + l16][ka],
                    *(const v8h*)&Wl[wn + 16 + l16][ka + 16]);

    // ---- D += hi*hi + hi*lo + lo*hi  (lo*lo ~ 2^-22, dropped) ------------
    acc[0][0] = wmma32(ah0, bh0, acc[0][0]);
    acc[0][0] = wmma32(ah0, bl0, acc[0][0]);
    acc[0][0] = wmma32(al0, bh0, acc[0][0]);
    acc[0][1] = wmma32(ah0, bh1, acc[0][1]);
    acc[0][1] = wmma32(ah0, bl1, acc[0][1]);
    acc[0][1] = wmma32(al0, bh1, acc[0][1]);
    acc[1][0] = wmma32(ah1, bh0, acc[1][0]);
    acc[1][0] = wmma32(ah1, bl0, acc[1][0]);
    acc[1][0] = wmma32(al1, bh0, acc[1][0]);
    acc[1][1] = wmma32(ah1, bh1, acc[1][1]);
    acc[1][1] = wmma32(ah1, bl1, acc[1][1]);
    acc[1][1] = wmma32(al1, bh1, acc[1][1]);
    __syncthreads();
  }

#pragma unroll
  for (int i = 0; i < 2; ++i)
#pragma unroll
    for (int j = 0; j < 2; ++j) {
      const int mf = m0 + wm + i * 16 + lh * 8;
      const int nf = n0 + wn + j * 16 + l16;
      const float bv = bias[nf];
      float* Cp = C + (size_t)mf * N + nf;
#pragma unroll
      for (int r = 0; r < 8; ++r) Cp[(size_t)r * N] = acc[i][j][r] + bv;
    }
}

// ---------------------------------------------------------------------------
// Partial T_g = K_g^T @ V_g over one S-slice of 256 rows (exact f32 WMMA).
// Grid: (32 head-blocks, 8 slices), block 256 = 8 waves.
// ---------------------------------------------------------------------------
__global__ __launch_bounds__(256) void kv_outer_kernel(
    const float* __restrict__ Klin, const float* __restrict__ Vlin,
    float* __restrict__ Tpart) {
  __shared__ __align__(16) float Ks[64][72];  // [s within chunk][d]
  __shared__ __align__(16) float Vs[64][72];

  const int tid  = threadIdx.x;
  const int lane = tid & 31;
  const int wave = tid >> 5;
  const int lh   = lane >> 4;
  const int l16  = lane & 15;
  const int g    = blockIdx.x;
  const int sl   = blockIdx.y;

  const float* Kg = Klin + (size_t)g * (2048 * 64) + (size_t)sl * 256 * 64;
  const float* Vg = Vlin + (size_t)g * (2048 * 64) + (size_t)sl * 256 * 64;

  const int wm  = (wave & 3) * 16;   // M (dk) tile
  const int wn0 = (wave >> 2) * 32;  // N (dv) tiles: wn0, wn0+16
  v8f acc[2] = {};

  for (int sb = 0; sb < 256; sb += 64) {
#pragma unroll
    for (int i = 0; i < 4; ++i) {
      const int idx = tid + i * 256;  // 1024 float4 per matrix
      const int r = idx >> 4, c = (idx & 15) * 4;
      *(float4*)&Ks[r][c] = *(const float4*)(Kg + (size_t)(sb + r) * 64 + c);
      *(float4*)&Vs[r][c] = *(const float4*)(Vg + (size_t)(sb + r) * 64 + c);
    }
    __syncthreads();
#pragma unroll
    for (int kk = 0; kk < 64; kk += 4) {
      const int kA = kk + 2 * lh;  // s index pair within chunk
      v2f a, b0, b1;
      a.x  = Ks[kA][wm + l16];       a.y  = Ks[kA + 1][wm + l16];
      b0.x = Vs[kA][wn0 + l16];      b0.y = Vs[kA + 1][wn0 + l16];
      b1.x = Vs[kA][wn0 + 16 + l16]; b1.y = Vs[kA + 1][wn0 + 16 + l16];
      acc[0] = wmma4(a, b0, acc[0]);
      acc[1] = wmma4(a, b1, acc[1]);
    }
    __syncthreads();
  }

  float* Tp = Tpart + ((size_t)g * 8 + sl) * (64 * 64);
#pragma unroll
  for (int j = 0; j < 2; ++j) {
    const int mf = wm + lh * 8;
    const int nf = wn0 + j * 16 + l16;
    float* p = Tp + (size_t)mf * 64 + nf;
#pragma unroll
    for (int r = 0; r < 8; ++r) p[r * 64] = acc[j][r];
  }
}

// Reduce 8 split-K partials; fold in norm = 1/sqrt(d_model) = 1/32.
__global__ void reduce_t_kernel(const float* __restrict__ Tpart,
                                float* __restrict__ T) {
  const int g = blockIdx.x;
  for (int e = threadIdx.x; e < 64 * 64; e += blockDim.x) {
    float s = 0.f;
#pragma unroll
    for (int p = 0; p < 8; ++p) s += Tpart[((size_t)g * 8 + p) * 4096 + e];
    T[(size_t)g * 4096 + e] = s * 0.03125f;
  }
}

// ---------------------------------------------------------------------------
// Attn_g[2048,64] = Q_g[2048,64] @ T_g[64,64]  (exact f32 WMMA).
// Grid: (32 head-blocks, 16 M-chunks of 128), block 256 = 8 waves.
// ---------------------------------------------------------------------------
__global__ __launch_bounds__(256) void qt_gemm_kernel(
    const float* __restrict__ Qlin, const float* __restrict__ T,
    float* __restrict__ Attn) {
  __shared__ __align__(16) float Qs[128][72];
  __shared__ __align__(16) float Ts[64][72];

  const int tid  = threadIdx.x;
  const int lane = tid & 31;
  const int wave = tid >> 5;
  const int lh   = lane >> 4;
  const int l16  = lane & 15;
  const int g    = blockIdx.x;
  const int mb   = blockIdx.y;

  const float* Qg = Qlin + (size_t)g * (2048 * 64) + (size_t)mb * 128 * 64;
  const float* Tg = T + (size_t)g * 4096;
  float* Og = Attn + (size_t)g * (2048 * 64) + (size_t)mb * 128 * 64;

  const int wm = (wave & 3) * 32;
  const int wn = (wave >> 2) * 32;

#pragma unroll
  for (int i = 0; i < 8; ++i) {  // Q tile: 2048 float4, 8/thread
    const int idx = tid + i * 256;
    const int r = idx >> 4, c = (idx & 15) * 4;
    *(float4*)&Qs[r][c] = *(const float4*)(Qg + (size_t)r * 64 + c);
  }
#pragma unroll
  for (int i = 0; i < 4; ++i) {  // T: 1024 float4, 4/thread
    const int idx = tid + i * 256;
    const int r = idx >> 4, c = (idx & 15) * 4;
    *(float4*)&Ts[r][c] = *(const float4*)(Tg + (size_t)r * 64 + c);
  }
  __syncthreads();

  v8f acc[2][2] = {};
#pragma unroll
  for (int kk = 0; kk < 64; kk += 4) {
    const int kA = kk + 2 * lh;
    v2f a0 = *(const v2f*)&Qs[wm + l16][kA];
    v2f a1 = *(const v2f*)&Qs[wm + 16 + l16][kA];
    v2f b0, b1;
    b0.x = Ts[kA][wn + l16];      b0.y = Ts[kA + 1][wn + l16];
    b1.x = Ts[kA][wn + 16 + l16]; b1.y = Ts[kA + 1][wn + 16 + l16];
    acc[0][0] = wmma4(a0, b0, acc[0][0]);
    acc[0][1] = wmma4(a0, b1, acc[0][1]);
    acc[1][0] = wmma4(a1, b0, acc[1][0]);
    acc[1][1] = wmma4(a1, b1, acc[1][1]);
  }

#pragma unroll
  for (int i = 0; i < 2; ++i)
#pragma unroll
    for (int j = 0; j < 2; ++j) {
      const int mf = wm + i * 16 + lh * 8;
      const int nf = wn + j * 16 + l16;
      float* p = Og + (size_t)mf * 64 + nf;
#pragma unroll
      for (int r = 0; r < 8; ++r) p[r * 64] = acc[i][j][r];
    }
}

// ---------------------------------------------------------------------------
extern "C" void kernel_launch(void* const* d_in, const int* in_sizes, int n_in,
                              void* d_out, int out_size, void* d_ws,
                              size_t ws_size, hipStream_t stream) {
  (void)in_sizes; (void)n_in; (void)out_size; (void)ws_size;

  const float* x   = (const float*)d_in[0];
  const float* y   = (const float*)d_in[1];
  const float* q_w = (const float*)d_in[2];
  const float* q_b = (const float*)d_in[3];
  const float* k_w = (const float*)d_in[4];
  const float* k_b = (const float*)d_in[5];
  const float* v_w = (const float*)d_in[6];
  const float* v_b = (const float*)d_in[7];
  const float* o_w = (const float*)d_in[8];
  const float* o_b = (const float*)d_in[9];
  float* out = (float*)d_out;

  char* w = (char*)d_ws;
  const size_t SZ = (size_t)4096 * 1024 * sizeof(float);  // 16 MB
  float* Qlin  = (float*)(w);
  float* Klin  = (float*)(w + SZ);
  float* Vlin  = (float*)(w + 2 * SZ);
  float* Attn  = (float*)(w + 3 * SZ);
  float* Tpart = (float*)(w + 4 * SZ);
  float* Tm    = (float*)(w + 4 * SZ + (size_t)32 * 8 * 4096 * sizeof(float));

  const dim3 blk(256);
  const dim3 gA(32, 16);  // 4096/128 x 1024/64

  gemm_bias_bt_f16x2_kernel<<<gA, blk, 0, stream>>>(x, q_w, q_b, Qlin, 1024, 1024);
  gemm_bias_bt_f16x2_kernel<<<gA, blk, 0, stream>>>(x, k_w, k_b, Klin, 1024, 1024);
  gemm_bias_bt_f16x2_kernel<<<gA, blk, 0, stream>>>(y, v_w, v_b, Vlin, 1024, 1024);
  kv_outer_kernel<<<dim3(32, 8), blk, 0, stream>>>(Klin, Vlin, Tpart);
  reduce_t_kernel<<<dim3(32), blk, 0, stream>>>(Tpart, Tm);
  qt_gemm_kernel<<<dim3(32, 16), blk, 0, stream>>>(Qlin, Tm, Attn);
  gemm_bias_bt_f16x2_kernel<<<gA, blk, 0, stream>>>(Attn, o_w, o_b, out, 1024, 1024);
}
